// UnivNetGenerator_9345848836766
// MI455X (gfx1250) — matrix operand order repaired and checked by reading, not compiled, not run on hardware
//
#include <hip/hip_runtime.h>
#include <hip/hip_bf16.h>
#include <cstdint>
#include <cstddef>

// ---------------------------------------------------------------------------
// CDNA5 (gfx1250) wave32 WMMA types
// ---------------------------------------------------------------------------
typedef __attribute__((ext_vector_type(16))) __bf16 v16bf;
typedef __attribute__((ext_vector_type(8)))  __bf16 v8bf;
typedef __attribute__((ext_vector_type(8)))  float  v8f;

// Async global->LDS DMA (ASYNCcnt path), guarded so compile never breaks.
// Builtin prototype (from hipcc diagnostic): takes generic `v4i*` pointers;
// clang infers global/LDS address spaces from pointer provenance.
#if defined(__gfx1250__) && __has_builtin(__builtin_amdgcn_global_load_async_to_lds_b128) && \
    __has_builtin(__builtin_amdgcn_s_wait_asynccnt)
#define HAVE_ASYNC_LDS 1
typedef int v4i __attribute__((vector_size(16)));
#else
#define HAVE_ASYNC_LDS 0
#endif

__device__ __forceinline__ __bf16 f2bf(float f) {
  unsigned u = __builtin_bit_cast(unsigned, f);
  unsigned r = (u + 0x7FFFu + ((u >> 16) & 1u)) >> 16;   // round-to-nearest-even
  return __builtin_bit_cast(__bf16, (unsigned short)r);
}
__device__ __forceinline__ float lrelu_f(float v) { return v >= 0.f ? v : 0.2f * v; }
__device__ __forceinline__ __bf16 bfz() { return __builtin_bit_cast(__bf16, (unsigned short)0); }

__device__ __forceinline__ v16bf combine(v8bf lo, v8bf hi) {
  return __builtin_shufflevector(lo, hi, 0,1,2,3,4,5,6,7,8,9,10,11,12,13,14,15);
}

// branch-free activations: v_exp_f32 + v_rcp_f32, no libm piecewise branches
__device__ __forceinline__ float fast_sigmoid(float x) {
  return __builtin_amdgcn_rcpf(1.f + __expf(-x));
}
__device__ __forceinline__ float fast_tanh(float x) {
  x = fminf(15.f, fmaxf(-15.f, x));
  float e = __expf(2.f * x);
  return (e - 1.f) * __builtin_amdgcn_rcpf(e + 1.f);
}

// ---------------------------------------------------------------------------
// Vectorized f32 -> bf16 cast (optional fused LeakyReLU). n % 8 == 0 fast path.
// ---------------------------------------------------------------------------
__global__ void cast_bf16_kernel(const float* __restrict__ src, __bf16* __restrict__ dst,
                                 size_t n, int act) {
  size_t i = ((size_t)blockIdx.x * blockDim.x + threadIdx.x) * 8;
  if (i + 8 <= n) {
    float4 f0 = *(const float4*)(src + i);
    float4 f1 = *(const float4*)(src + i + 4);
    if (act) {
      f0.x = lrelu_f(f0.x); f0.y = lrelu_f(f0.y); f0.z = lrelu_f(f0.z); f0.w = lrelu_f(f0.w);
      f1.x = lrelu_f(f1.x); f1.y = lrelu_f(f1.y); f1.z = lrelu_f(f1.z); f1.w = lrelu_f(f1.w);
    }
    union { __bf16 h[8]; uint4 v; } o;
    o.h[0] = f2bf(f0.x); o.h[1] = f2bf(f0.y); o.h[2] = f2bf(f0.z); o.h[3] = f2bf(f0.w);
    o.h[4] = f2bf(f1.x); o.h[5] = f2bf(f1.y); o.h[6] = f2bf(f1.z); o.h[7] = f2bf(f1.w);
    *(uint4*)(dst + i) = o.v;
  } else {
    for (size_t j = i; j < n; ++j) {
      float v = src[j];
      if (act) v = lrelu_f(v);
      dst[j] = f2bf(v);
    }
  }
}

// ---------------------------------------------------------------------------
// Weight cast + transpose: src (Cout, Cin, K) f32 -> dst [k][Cout][Cpad] bf16,
// zero-padded channels so the GEMM needs no reduction-tail guards.
// ---------------------------------------------------------------------------
__global__ void wcast_tr(const float* __restrict__ src, __bf16* __restrict__ dst,
                         int Cout, int Cin, int K, int Cpad) {
  int i = blockIdx.x * blockDim.x + threadIdx.x;
  int total = K * Cout * Cpad;
  if (i >= total) return;
  int ci = i % Cpad;
  int co = (i / Cpad) % Cout;
  int k  = i / (Cpad * Cout);
  dst[i] = (ci < Cin) ? f2bf(src[((size_t)co * Cin + ci) * K + k]) : bfz();
}

// ---------------------------------------------------------------------------
// Conv1d as per-tap WMMA GEMM:  out = sum_k W_k (Cout x Cpad) * X_k (Cpad x N)
//  - input slab staged channel-minor in LDS -> B fragments = 2x ds_load_b128
//  - weights pre-transposed [k][Cout][Cpad]  -> A fragments = 2x global b128,
//    reused across NT n-tiles.
//  - input may be bf16 (in) or f32 (inf, with fused LeakyReLU): the f32 path
//    removes whole-tensor cast passes for the big dilated convs.
// mode bit0: outb written in LVC-permuted layout [b][l][layer][k][o][ci]
// mode bit1: outf written transposed (b, l, co)
// ---------------------------------------------------------------------------
template <int K, int NT>
__global__ void __launch_bounds__(128)
conv1d_wmma(const __bf16* __restrict__ in, const float* __restrict__ inf, int act_in,
            const __bf16* __restrict__ w2,
            const float* __restrict__ bias, const float* __restrict__ res,
            float* __restrict__ outf, __bf16* __restrict__ outb,
            int B, int Cin, int Cpad, int Cout, int L,
            int pad, int dil, int act, int mode) {
  const int nwaves = blockDim.x >> 5;
  const int l0 = blockIdx.x * (16 * NT);
  const int b  = blockIdx.z;
  const int width  = 16 * NT + (K - 1) * dil;
  const int lstart = l0 - pad * dil;

  __shared__ __align__(16) __bf16 slab[8192];   // [pos][Cpad]
  for (int idx = threadIdx.x; idx < Cpad * width; idx += blockDim.x) {
    int ci = idx / width, j = idx % width;      // ci-major iteration: coalesced global
    int l = lstart + j;
    __bf16 e = bfz();
    if (ci < Cin && l >= 0 && l < L) {
      if (inf) {                                // fused f32 read + lrelu + cast
        float v = inf[((size_t)b * Cin + ci) * L + l];
        if (act_in) v = lrelu_f(v);
        e = f2bf(v);
      } else {
        e = in[((size_t)b * Cin + ci) * L + l];
      }
    }
    slab[j * Cpad + ci] = e;
  }
  __syncthreads();

  const int wave = threadIdx.x >> 5;
  const int lane = threadIdx.x & 31;
  const int half = lane >> 4;
  const int n    = lane & 15;
  const int co0  = (blockIdx.y * nwaves + wave) * 16;
  if (co0 >= Cout) return;
  const int co = co0 + n;                       // A row (same for both lane halves)

  v8f acc[NT];
#pragma unroll
  for (int nt = 0; nt < NT; ++nt) acc[nt] = v8f{0.f,0.f,0.f,0.f,0.f,0.f,0.f,0.f};

#pragma unroll
  for (int k = 0; k < K; ++k) {
    for (int c0 = 0; c0 < Cpad; c0 += 32) {
      const __bf16* ar = w2 + ((size_t)(k * Cout + co)) * Cpad + c0 + half * 8;
      __builtin_prefetch((const void*)(ar + Cpad), 0, 1);
      v16bf a = combine(*(const v8bf*)ar, *(const v8bf*)(ar + 16));
#pragma unroll
      for (int nt = 0; nt < NT; ++nt) {
        const __bf16* br = slab + (nt * 16 + n + k * dil) * Cpad + c0 + half * 8;
        v16bf bf = combine(*(const v8bf*)br, *(const v8bf*)(br + 16));
        acc[nt] = __builtin_amdgcn_wmma_f32_16x16x32_bf16(false, a, false, bf, (short)0,
                                                          acc[nt], false, false);
      }
    }
  }

#pragma unroll
  for (int nt = 0; nt < NT; ++nt) {
#pragma unroll
    for (int r = 0; r < 8; ++r) {
      int M  = r + half * 8;
      int cc = co0 + M;
      int l  = l0 + nt * 16 + n;
      float v = acc[nt][r] + bias[cc];
      if (act) v = lrelu_f(v);
      size_t gnorm = ((size_t)b * Cout + cc) * L + l;
      if (res) v += res[gnorm];
      if (outf) {
        size_t go = (mode & 2) ? (((size_t)b * L + l) * Cout + cc) : gnorm;
        outf[go] = v;
      }
      if (outb) {
        size_t go;
        if (mode & 1) {   // permuted dynamic-kernel layout [b][l][layer][k][o][ci]
          int kq = cc % 3, o = (cc / 3) & 63, ci = (cc / 192) & 31, layer = cc / 6144;
          go = ((size_t)b * L + l) * (size_t)Cout + (size_t)layer * 6144 + (kq * 64 + o) * 32 + ci;
        } else {
          go = gnorm;
        }
        outb[go] = f2bf(v);
      }
    }
  }
}

// ---------------------------------------------------------------------------
// Location-variable convolution + gated activation, WMMA.
// kernP: permuted layout [b][l][layer][k][o][ci]  (contiguous 12KB per (b,l,layer))
// biasT: transposed (b, l, 256)
// Work item = (n-tile, m-pair). M-pair p covers M-tiles {p, p+2}, i.e. exactly
// the sigmoid/tanh channel pair (c, c+32): only 2 accumulators + 6 A-frags
// live per item -> no VGPR spills.  Slab is zero-padded, so no B-tile mask.
// Kernel slice staged via async global->LDS DMA when available (ASYNCcnt).
// ---------------------------------------------------------------------------
__global__ void __launch_bounds__(256)
lvc_wmma(const __bf16* __restrict__ y, const __bf16* __restrict__ kernP,
         const float* __restrict__ biasT, float* __restrict__ x,
         int B, int T, int L, int hop, int layer) {
  const int l = blockIdx.x;
  const int b = blockIdx.y;
  const int t0 = l * hop;
  const int Ws = (hop < 16) ? (hop + 18) : (hop + 2);  // zero-pad staging when hop < tile

  __shared__ __align__(16) __bf16 karr[6144];      // [k][o][ci]
  __shared__ __align__(16) __bf16 slab[32 * 258];  // [pos][ci]

  {  // contiguous 12KB kernel-slice copy, 16B vectors
    const uint4* src = (const uint4*)(kernP + ((size_t)b * L + l) * 24576 + (size_t)layer * 6144);
    uint4* dst = (uint4*)karr;
#if HAVE_ASYNC_LDS
    for (int i = threadIdx.x; i < 768; i += blockDim.x)
      __builtin_amdgcn_global_load_async_to_lds_b128(
          (v4i*)(src + i), (v4i*)(dst + i), 0, 0);
#else
    for (int i = threadIdx.x; i < 768; i += blockDim.x) dst[i] = src[i];
#endif
  }
  for (int idx = threadIdx.x; idx < 32 * Ws; idx += blockDim.x) {
    int ci = idx / Ws, j = idx % Ws;               // ci-major: coalesced global
    int t = t0 + j - 1;
    slab[j * 32 + ci] = (t >= 0 && t < T) ? y[((size_t)b * 32 + ci) * T + t] : bfz();
  }
#if HAVE_ASYNC_LDS
  __builtin_amdgcn_s_wait_asynccnt(0);
#endif
  __syncthreads();

  const int wave = threadIdx.x >> 5, nwaves = blockDim.x >> 5;
  const int lane = threadIdx.x & 31;
  const int half = lane >> 4;
  const int n    = lane & 15;
  const int nitems = ((hop + 15) / 16) * 2;
  const float* bb = biasT + ((size_t)b * L + l) * 256 + layer * 64;

  for (int item = wave; item < nitems; item += nwaves) {
    const int nt = item >> 1;
    const int p  = item & 1;        // m-pair: M-tiles {p, p+2} = channels c, c+32
    const int h0 = nt * 16;
    v8f accS = v8f{0.f,0.f,0.f,0.f,0.f,0.f,0.f,0.f};
    v8f accT = v8f{0.f,0.f,0.f,0.f,0.f,0.f,0.f,0.f};

#pragma unroll
    for (int kk = 0; kk < 3; ++kk) {
      const __bf16* br = slab + (h0 + n + kk) * 32 + half * 8;
      v16bf bf = combine(*(const v8bf*)br, *(const v8bf*)(br + 16));
      const __bf16* ar = karr + (kk * 64 + p * 16 + n) * 32 + half * 8;
      v16bf aS = combine(*(const v8bf*)ar, *(const v8bf*)(ar + 16));
      accS = __builtin_amdgcn_wmma_f32_16x16x32_bf16(false, aS, false, bf, (short)0,
                                                     accS, false, false);
      v16bf aT = combine(*(const v8bf*)(ar + 1024), *(const v8bf*)(ar + 1040));
      accT = __builtin_amdgcn_wmma_f32_16x16x32_bf16(false, aT, false, bf, (short)0,
                                                     accT, false, false);
    }

    // fused gating epilogue: x += sigmoid(accS + bias_c) * tanh(accT + bias_{c+32})
#pragma unroll
    for (int r = 0; r < 8; ++r) {
      int M = r + half * 8;
      int h = h0 + n;
      if (h < hop) {
        int t = t0 + h;
        int c = p * 16 + M;
        float s  = fast_sigmoid(accS[r] + bb[c]);
        float tv = fast_tanh(accT[r] + bb[32 + c]);
        x[((size_t)b * 32 + c) * T + t] += s * tv;
      }
    }
  }
}

// ---------------------------------------------------------------------------
// ConvTranspose1d (stride s, kernel 2s, padding s/2), fused input LeakyReLU.
// ---------------------------------------------------------------------------
__global__ void convT_valu(const float* __restrict__ xin, const float* __restrict__ wt,
                           const float* __restrict__ bias, float* __restrict__ xout,
                           int B, int C, int Tin, int s) {
  const int Tout = Tin * s;
  size_t i = (size_t)blockIdx.x * blockDim.x + threadIdx.x;
  if (i >= (size_t)B * C * Tout) return;
  int tp = (int)(i % Tout);
  int o  = (int)((i / Tout) % C);
  int b  = (int)(i / ((size_t)Tout * C));

  const int K2 = 2 * s;
  const int P  = K2 - 1 - s / 2;
  int k0 = (P - tp) % s; if (k0 < 0) k0 += s;

  float acc = bias[o];
  for (int kk = k0; kk < K2; kk += s) {
    int num = tp + kk - P;
    if (num < 0 || (num % s) != 0) continue;
    int t = num / s;
    if (t >= Tin) continue;
    int kf = K2 - 1 - kk;
    for (int ci = 0; ci < C; ++ci) {
      float xv = lrelu_f(xin[((size_t)b * C + ci) * Tin + t]);
      acc += wt[((size_t)ci * C + o) * K2 + kf] * xv;
    }
  }
  xout[i] = acc;
}

// ---------------------------------------------------------------------------
// Pre conv: reflect-pad(3) on z (B,Cin,L) then conv (Cout,Cin,7) + bias.
// ---------------------------------------------------------------------------
__global__ void pre_conv(const float* __restrict__ z, const float* __restrict__ w,
                         const float* __restrict__ bias, float* __restrict__ x,
                         int B, int Cin, int Cout, int L) {
  int i = blockIdx.x * blockDim.x + threadIdx.x;
  if (i >= B * Cout * L) return;
  int l = i % L, o = (i / L) % Cout, b = i / (L * Cout);
  float acc = bias[o];
  for (int k = 0; k < 7; ++k) {
    int t = l + k - 3;
    if (t < 0) t = -t;
    if (t >= L) t = 2 * L - 2 - t;
    for (int ci = 0; ci < Cin; ++ci)
      acc += w[(o * Cin + ci) * 7 + k] * z[((size_t)b * Cin + ci) * L + t];
  }
  x[i] = acc;
}

// ---------------------------------------------------------------------------
// Post: lrelu(x) -> reflect-pad(3) -> conv (1,C,7)+b -> tanh
// ---------------------------------------------------------------------------
__global__ void post_conv(const float* __restrict__ x, const float* __restrict__ w,
                          const float* __restrict__ b0, float* __restrict__ out,
                          int B, int C, int T) {
  size_t i = (size_t)blockIdx.x * blockDim.x + threadIdx.x;
  if (i >= (size_t)B * T) return;
  int t = (int)(i % T);
  int b = (int)(i / T);
  float acc = b0[0];
  for (int k = 0; k < 7; ++k) {
    int tt = t + k - 3;
    if (tt < 0) tt = -tt;
    if (tt >= T) tt = 2 * T - 2 - tt;
    for (int ci = 0; ci < C; ++ci)
      acc += w[ci * 7 + k] * lrelu_f(x[((size_t)b * C + ci) * T + tt]);
  }
  out[i] = fast_tanh(acc);
}

// ---------------------------------------------------------------------------
// Host orchestration
// ---------------------------------------------------------------------------
extern "C" void kernel_launch(void* const* d_in, const int* in_sizes, int n_in,
                              void* d_out, int out_size, void* d_ws, size_t ws_size,
                              hipStream_t stream) {
  (void)in_sizes; (void)n_in; (void)out_size; (void)ws_size;

  // d_in[0]=c, d_in[1]=z, then params jax-tree (dicts in sorted key order).
  // Per-block tensor offsets:
  //  +0..3 cb[0..3], +4 ct_b, +5 ct_w, +6..9 cw[0..3],
  //  +10 b_b, +11 b_w, +12 in_b, +13 in_w, +14 k_b, +15 k_w,
  //  +16.. res[j] = {rw1, rb1, rw2, rb2}
  const float* c = (const float*)d_in[0];
  const float* z = (const float*)d_in[1];
  auto F = [&](int i) -> const float* { return (const float*)d_in[i]; };

  const int B = 8, L = 256, CH = 32;
  const int strides[3] = {8, 8, 4};
  const int hops[3]    = {8, 64, 256};
  const int dils[4]    = {1, 3, 9, 27};
  const int blkbase[3] = {2, 30, 58};
  const size_t Tmax = 65536;

  // ---- workspace arena ----
  char* arena = (char*)d_ws;
  size_t off = 0;
  auto alloc = [&](size_t bytes) -> void* {
    void* p = arena + off;
    off += (bytes + 255) & ~(size_t)255;
    return p;
  };
  float*  bufA  = (float*) alloc((size_t)B * CH * Tmax * 4);
  float*  bufB  = (float*) alloc((size_t)B * CH * Tmax * 4);
  __bf16* ybf   = (__bf16*)alloc((size_t)B * CH * Tmax * 2);
  __bf16* cbf   = (__bf16*)alloc((size_t)B * 100 * L * 2);
  float*  hf    = (float*) alloc((size_t)B * 64 * L * 4);
  float*  h2f   = (float*) alloc((size_t)B * 64 * L * 4);
  __bf16* hb    = (__bf16*)alloc((size_t)B * 64 * L * 2);
  __bf16* h2b   = (__bf16*)alloc((size_t)B * 64 * L * 2);
  __bf16* r1b   = (__bf16*)alloc((size_t)B * 64 * L * 2);
  __bf16* kern  = (__bf16*)alloc((size_t)B * 24576 * L * 2);   // permuted [b][l][...]
  float*  biasT = (float*) alloc((size_t)B * L * 256 * 4);     // transposed (b,l,ch)
  __bf16* wpool = (__bf16*)alloc((size_t)5000192 * 2);

  // weight-pool element offsets (identical layout every block)
  size_t wo = 0;
  auto wtake = [&](size_t elems) -> size_t { size_t o = wo; wo += (elems + 127) & ~(size_t)127; return o; };
  size_t wo_in = wtake((size_t)5 * 64 * 128);
  size_t wo_rw[6]; for (int j = 0; j < 6; ++j) wo_rw[j] = wtake((size_t)3 * 64 * 64);
  size_t wo_kw = wtake((size_t)3 * 24576 * 64);
  size_t wo_bw = wtake((size_t)3 * 256 * 64);
  size_t wo_cw[4]; for (int j = 0; j < 4; ++j) wo_cw[j] = wtake((size_t)3 * 32 * 32);

  auto cast = [&](const float* s, __bf16* d, size_t n, int act) {
    unsigned blocks = (unsigned)((n / 8 + 255) / 256);
    cast_bf16_kernel<<<dim3(blocks), dim3(256), 0, stream>>>(s, d, n, act);
  };
  auto wtr = [&](const float* s, __bf16* d, int Cout, int Cin, int K, int Cpad) {
    int total = K * Cout * Cpad;
    wcast_tr<<<dim3((total + 255) / 256), dim3(256), 0, stream>>>(s, d, Cout, Cin, K, Cpad);
  };
  auto conv3 = [&](const __bf16* in, const float* inf, int act_in,
                   const __bf16* w, const float* bias, const float* res,
                   float* outf, __bf16* outb, int Cin, int Cout, int Ln,
                   int pad, int d, int act, int mode) {
    int Cpad = (Cin + 31) & ~31;
    int waves = (Cout >= 64) ? 4 : 2;
    dim3 blk(32 * waves);
    dim3 grd(Ln / 64, (Cout + 16 * waves - 1) / (16 * waves), B);
    conv1d_wmma<3, 4><<<grd, blk, 0, stream>>>(in, inf, act_in, w, bias, res, outf, outb,
                                               B, Cin, Cpad, Cout, Ln, pad, d, act, mode);
  };

  // ---- pre: cast c; pre conv z -> x(256) ----
  cast(c, cbf, (size_t)B * 100 * L, 0);
  {
    int total = B * CH * L;
    pre_conv<<<dim3((total + 255) / 256), dim3(256), 0, stream>>>(
        z, F(89), F(88), bufA, B, 64, CH, L);
  }

  int Tin = 256;
  for (int bi = 0; bi < 3; ++bi) {
    const int base = blkbase[bi];
    const int s = strides[bi], hop = hops[bi];
    const int Tout = Tin * s;
    float* xin  = (bi & 1) ? bufB : bufA;
    float* xout = (bi & 1) ? bufA : bufB;

    // weights -> transposed bf16 pool
    wtr(F(base + 13), wpool + wo_in, 64, 100, 5, 128);
    for (int j = 0; j < 3; ++j) {
      wtr(F(base + 16 + 4 * j),     wpool + wo_rw[2 * j],     64, 64, 3, 64);
      wtr(F(base + 16 + 4 * j + 2), wpool + wo_rw[2 * j + 1], 64, 64, 3, 64);
    }
    wtr(F(base + 15), wpool + wo_kw, 24576, 64, 3, 64);
    wtr(F(base + 11), wpool + wo_bw, 256, 64, 3, 64);
    for (int j = 0; j < 4; ++j) wtr(F(base + 6 + j), wpool + wo_cw[j], 32, 32, 3, 32);

    // upsample: x = convT(lrelu(x))
    {
      size_t total = (size_t)B * CH * Tout;
      convT_valu<<<dim3((unsigned)((total + 255) / 256)), dim3(256), 0, stream>>>(
          xin, F(base + 5), F(base + 4), xout, B, CH, Tin, s);
    }

    // kernel predictor (all WMMA)
    {
      dim3 blk(128), grd(L / 32, 1, B);
      conv1d_wmma<5, 2><<<grd, blk, 0, stream>>>(cbf, nullptr, 0, wpool + wo_in,
                                                 F(base + 12), nullptr,
                                                 hf, hb, B, 100, 128, 64, L, 2, 1, 1, 0);
    }
    float* hF = hf; float* h2F = h2f; __bf16* hB = hb; __bf16* h2B = h2b;
    for (int j = 0; j < 3; ++j) {
      conv3(hB,  nullptr, 0, wpool + wo_rw[2 * j],     F(base + 17 + 4 * j), nullptr,
            nullptr, r1b, 64, 64, L, 1, 1, 1, 0);
      conv3(r1b, nullptr, 0, wpool + wo_rw[2 * j + 1], F(base + 19 + 4 * j), hF,
            h2F, h2B, 64, 64, L, 1, 1, 1, 0);
      float* tf = hF; hF = h2F; h2F = tf;
      __bf16* tb = hB; hB = h2B; h2B = tb;
    }
    conv3(hB, nullptr, 0, wpool + wo_kw, F(base + 14), nullptr,
          nullptr, kern, 64, 24576, L, 1, 1, 0, 1);
    conv3(hB, nullptr, 0, wpool + wo_bw, F(base + 10), nullptr,
          biasT, nullptr, 64, 256, L, 1, 1, 0, 2);

    // 4 dilated convs + LVC; conv reads f32 x with fused lrelu+cast (no x casts)
    for (int i = 0; i < 4; ++i) {
      const int d = dils[i];
      conv3(nullptr, xout, 1, wpool + wo_cw[i], F(base + i), nullptr,
            nullptr, ybf, CH, CH, Tout, d, d, 1, 0);
      lvc_wmma<<<dim3(L, B), dim3(256), 0, stream>>>(ybf, kern, biasT, xout,
                                                     B, Tout, L, hop, i);
    }
    Tin = Tout;
  }

  // ---- post: final x is in bufB (bi=2 wrote xout=bufB) ----
  {
    size_t total = (size_t)B * Tin;
    post_conv<<<dim3((unsigned)((total + 255) / 256)), dim3(256), 0, stream>>>(
        bufB, F(87), F(86), (float*)d_out, B, CH, Tin);
  }
}
